// MDA_28656021799306
// MI455X (gfx1250) — matrix operand order repaired and checked by reading, not compiled, not run on hardware
//
#include <hip/hip_runtime.h>

// ---------------- problem constants ----------------
#define N_NODES 2060
#define F_OUT   901
#define M_ROWS  1778
#define E_EDGES (N_NODES * 64)          // 131840
#define TOT_E   (E_EDGES + N_NODES)     // + self loops
#define KPAD    928                     // 901 padded to multiple of 32
#define K1      (2 * KPAD)              // 1856 : padded concat-K for layer 1
#define N1      1024
#define N2      512
#define N3      64
#define NTRAIN  100000
#define NTEST   30000
#define BTOT    (NTRAIN + NTEST)        // 130000 = 8125 * 16 exactly
#define NB_PART 240
#define BN_EPS  1e-5f

// GAT GEMM tiling
#define G_MP    2064                    // M padded (129 tiles)
#define G_KP    2080                    // K padded (65 steps of 32)
#define G_KS    65
#define G_NT    57                      // ceil(901/16)
// MLP tiling
#define L1_KS   (K1 / 32)               // 58
#define L1_NT   (N1 / 16)               // 64
#define L2_KS   (N1 / 32)               // 32
#define L2_NT   (N2 / 16)               // 32
#define L3_KS   (N2 / 32)               // 16
#define L3_NT   (N3 / 16)               // 4

typedef _Float16 half_t;
typedef __attribute__((ext_vector_type(16))) _Float16 v16h;
typedef __attribute__((ext_vector_type(8)))  _Float16 v8h;
typedef __attribute__((ext_vector_type(8)))  float    v8f;

// ---------------- small device helpers ----------------
__device__ inline v8f zero8() {
  v8f z;
#pragma unroll
  for (int i = 0; i < 8; ++i) z[i] = 0.f;
  return z;
}

__device__ inline v8f wmma16(v16h a, v16h b, v8f c) {
  return __builtin_amdgcn_wmma_f32_16x16x32_f16(false, a, false, b, (short)0, c,
                                                false, false);
}

__device__ inline v16h join16(v8h lo, v8h hi) {
  v16h a;
#pragma unroll
  for (int j = 0; j < 8; ++j) { a[j] = lo[j]; a[8 + j] = hi[j]; }
  return a;
}

// monotone uint encoding of float for atomicMax-based segment max
__device__ inline unsigned encF(float v) {
  unsigned u = __float_as_uint(v);
  return (u >> 31) ? ~u : (u | 0x80000000u);
}
__device__ inline float decF(unsigned e) {
  return (e >> 31) ? __uint_as_float(e ^ 0x80000000u) : __uint_as_float(~e);
}

__device__ inline float block_reduce_sum(float v, float* red) {
  const int tid = threadIdx.x;
  red[tid] = v;
  __syncthreads();
  for (int off = 128; off > 0; off >>= 1) {
    if (tid < off) red[tid] += red[tid + off];
    __syncthreads();
  }
  float r = red[0];
  __syncthreads();
  return r;
}

__device__ inline int dv_row(int m) { return (m < 901) ? m : m + 282; } // 1183+(m-901)

// ---------------- prep: pad+convert A (x_feat) to f16 ----------------
__global__ void pack_xh_kernel(const float* __restrict__ X, half_t* __restrict__ Xh) {
  const long long total = (long long)G_MP * G_KP;
  for (long long t = blockIdx.x * (long long)blockDim.x + threadIdx.x; t < total;
       t += (long long)gridDim.x * blockDim.x) {
    int r = (int)(t / G_KP), k = (int)(t % G_KP);
    Xh[t] = (r < N_NODES && k < N_NODES) ? (half_t)X[(long long)r * N_NODES + k]
                                         : (half_t)0.f;
  }
}

// ---------------- prep: swizzle B matrices into WMMA fragment order ----------
// dst[(((nt*KS + ks)*32 + lane)*16 + j)] = B[k][n] with
//   k = ks*32 + (lane>>4)*8 + (j&7) + (j>>3)*16,  n = nt*16 + (lane&15)
// mode 0: direct (zero outside Ksrc x Nsrc)
// mode 1: W1 concat-padding remap (rows 0..900 -> 0..900, 928..1828 -> 901..1801)
__global__ void pack_b_kernel(const float* __restrict__ src, half_t* __restrict__ dst,
                              int Ksrc, int Nsrc, int KS, int NT, int mode) {
  const long long total = (long long)NT * KS * 32 * 16;
  for (long long t = blockIdx.x * (long long)blockDim.x + threadIdx.x; t < total;
       t += (long long)gridDim.x * blockDim.x) {
    int j = (int)(t & 15);
    int lane = (int)((t >> 4) & 31);
    long long rest = t >> 9;
    int ks = (int)(rest % KS);
    int nt = (int)(rest / KS);
    int k = ks * 32 + ((lane >> 4) * 8) + (j & 7) + ((j >> 3) * 16);
    int n = nt * 16 + (lane & 15);
    float v = 0.f;
    if (n < Nsrc) {
      int kk = -1;
      if (mode == 1) {
        if (k < 901) kk = k;
        else if (k >= KPAD && k < KPAD + 901) kk = k - 27;
      } else {
        if (k < Ksrc) kk = k;
      }
      if (kk >= 0 && kk < Ksrc) v = src[(long long)kk * Nsrc + n];
    }
    dst[t] = (half_t)v;
  }
}

// ---------------- GAT: H = x_feat @ Wg via WMMA ----------------
__global__ __launch_bounds__(256) void gat_gemm_kernel(const half_t* __restrict__ Xh,
                                                       const half_t* __restrict__ Bp,
                                                       float* __restrict__ H) {
  const int mt = blockIdx.x;              // 129 M-tiles
  const int wv = threadIdx.x >> 5;
  const int lane = threadIdx.x & 31;
  const int l16 = lane & 15, lh = lane >> 4;
  const half_t* arow = Xh + (long long)(mt * 16 + l16) * G_KP;
  for (int nt = wv; nt < G_NT; nt += 8) {
    const int n = nt * 16 + l16;
    const half_t* bp = Bp + (((long long)nt * G_KS) * 32 + lane) * 16;
    v8f acc = zero8();
    for (int ks = 0; ks < G_KS; ++ks) {
      const int off0 = ks * 32 + lh * 8;
      v8h lo = *(const v8h*)(arow + off0);
      v8h hi = *(const v8h*)(arow + off0 + 16);
      v16h a = join16(lo, hi);
      v16h b = *(const v16h*)bp;
      __builtin_prefetch(bp + 512, 0, 1);
      bp += 512;                          // next K-step fragment
      acc = wmma16(a, b, acc);
    }
#pragma unroll
    for (int r = 0; r < 8; ++r) {
      int mm = mt * 16 + r + lh * 8;
      if (mm < N_NODES && n < F_OUT) H[(long long)mm * F_OUT + n] = acc[r];
    }
  }
}

// a_s[i] = H[i]·att_src ; a_d[i] = H[i]·att_dst
__global__ __launch_bounds__(256) void attn_scores_kernel(const float* __restrict__ H,
                                                          const float* __restrict__ as_v,
                                                          const float* __restrict__ ad_v,
                                                          float* __restrict__ a_s,
                                                          float* __restrict__ a_d) {
  __shared__ float red[256];
  const int row = blockIdx.x;
  float s = 0.f, d = 0.f;
  for (int f = threadIdx.x; f < F_OUT; f += 256) {
    float h = H[(long long)row * F_OUT + f];
    s += h * as_v[f];
    d += h * ad_v[f];
  }
  float S = block_reduce_sum(s, red);
  float D = block_reduce_sum(d, red);
  if (threadIdx.x == 0) { a_s[row] = S; a_d[row] = D; }
}

__global__ void gat_init_kernel(unsigned* __restrict__ emax, float* __restrict__ denom,
                                float* __restrict__ gatout) {
  const long long total = (long long)N_NODES * F_OUT;
  for (long long t = blockIdx.x * (long long)blockDim.x + threadIdx.x; t < total;
       t += (long long)gridDim.x * blockDim.x) {
    gatout[t] = 0.f;
    if (t < N_NODES) { emax[t] = encF(-3.0e38f); denom[t] = 0.f; }
  }
}

__global__ void edge_logit_kernel(const int* __restrict__ ei, const float* __restrict__ a_s,
                                  const float* __restrict__ a_d, float* __restrict__ logit,
                                  unsigned* __restrict__ emax) {
  for (int e = blockIdx.x * blockDim.x + threadIdx.x; e < TOT_E;
       e += gridDim.x * blockDim.x) {
    int src, dst;
    if (e < E_EDGES) { src = ei[e]; dst = ei[E_EDGES + e]; }
    else             { src = dst = e - E_EDGES; }
    float v = a_s[src] + a_d[dst];
    v = (v >= 0.f) ? v : 0.2f * v;        // leaky_relu(0.2)
    logit[e] = v;
    atomicMax(&emax[dst], encF(v));
  }
}

__global__ void edge_exp_kernel(const int* __restrict__ ei, const float* __restrict__ logit,
                                const unsigned* __restrict__ emax, float* __restrict__ ex,
                                float* __restrict__ denom) {
  for (int e = blockIdx.x * blockDim.x + threadIdx.x; e < TOT_E;
       e += gridDim.x * blockDim.x) {
    int dst = (e < E_EDGES) ? ei[E_EDGES + e] : (e - E_EDGES);
    float x = __expf(logit[e] - decF(emax[dst]));
    ex[e] = x;
    atomicAdd(&denom[dst], x);
  }
}

__global__ void edge_alpha_kernel(const int* __restrict__ ei, const float* __restrict__ denom,
                                  float* __restrict__ ex) {
  for (int e = blockIdx.x * blockDim.x + threadIdx.x; e < TOT_E;
       e += gridDim.x * blockDim.x) {
    int dst = (e < E_EDGES) ? ei[E_EDGES + e] : (e - E_EDGES);
    ex[e] = ex[e] / denom[dst];
  }
}

__global__ void edge_scatter_kernel(const int* __restrict__ ei, const float* __restrict__ alpha,
                                    const float* __restrict__ H, float* __restrict__ gatout) {
  const long long total = (long long)TOT_E * F_OUT;
  for (long long t = blockIdx.x * (long long)blockDim.x + threadIdx.x; t < total;
       t += (long long)gridDim.x * blockDim.x) {
    int e = (int)(t / F_OUT), f = (int)(t % F_OUT);
    int src, dst;
    if (e < E_EDGES) { src = ei[e]; dst = ei[E_EDGES + e]; }
    else             { src = dst = e - E_EDGES; }
    atomicAdd(&gatout[(long long)dst * F_OUT + f], alpha[e] * H[(long long)src * F_OUT + f]);
  }
}

__global__ void gat_bias_relu_kernel(float* __restrict__ g, const float* __restrict__ gat_b) {
  const long long total = (long long)N_NODES * F_OUT;
  for (long long t = blockIdx.x * (long long)blockDim.x + threadIdx.x; t < total;
       t += (long long)gridDim.x * blockDim.x) {
    float v = g[t] + gat_b[t % F_OUT];
    g[t] = v > 0.f ? v : 0.f;
  }
}

// ---------------- MS_CAM: batch-norm statistics + fusion ----------------
__global__ __launch_bounds__(256) void t0_stats_kernel(const float* __restrict__ g,
                                                       const float* __restrict__ mirna,
                                                       const float* __restrict__ lw1,
                                                       const float* __restrict__ lb1,
                                                       float* __restrict__ t0,
                                                       float* __restrict__ part) {
  __shared__ float red[256];
  const float w0 = lw1[0], w1 = lw1[1], b = lb1[0];
  const long long total = (long long)M_ROWS * F_OUT;
  float s = 0.f, q = 0.f;
  for (long long t = blockIdx.x * (long long)blockDim.x + threadIdx.x; t < total;
       t += (long long)gridDim.x * blockDim.x) {
    int m = (int)(t / F_OUT), w = (int)(t % F_OUT);
    float x0 = g[(long long)dv_row(m) * F_OUT + w];
    float x1 = mirna[t];
    float v = w0 * x0 + w1 * x1 + b;
    t0[t] = v;
    s += v;
    q += v * v;
  }
  float S = block_reduce_sum(s, red);
  float Q = block_reduce_sum(q, red);
  if (threadIdx.x == 0) { part[blockIdx.x] = S; part[NB_PART + blockIdx.x] = Q; }
}

__global__ __launch_bounds__(256) void t2_stats_kernel(const float* __restrict__ t0,
                                                       const float* __restrict__ lw2,
                                                       const float* __restrict__ lb2,
                                                       const float* __restrict__ st0,
                                                       float* __restrict__ part) {
  __shared__ float red[256];
  const float mu0 = st0[0], rs0 = st0[1];
  const float w0 = lw2[0], w1c = lw2[1], b0 = lb2[0], b1c = lb2[1];
  const long long total = (long long)M_ROWS * F_OUT;
  float s0 = 0.f, s1 = 0.f, q0 = 0.f, q1 = 0.f;
  for (long long t = blockIdx.x * (long long)blockDim.x + threadIdx.x; t < total;
       t += (long long)gridDim.x * blockDim.x) {
    float r = (t0[t] - mu0) * rs0;
    r = fmaxf(r, 0.f);
    float v0 = w0 * r + b0, v1 = w1c * r + b1c;
    s0 += v0; q0 += v0 * v0;
    s1 += v1; q1 += v1 * v1;
  }
  float S0 = block_reduce_sum(s0, red);
  float S1 = block_reduce_sum(s1, red);
  float Q0 = block_reduce_sum(q0, red);
  float Q1 = block_reduce_sum(q1, red);
  if (threadIdx.x == 0) {
    part[0 * NB_PART + blockIdx.x] = S0;
    part[1 * NB_PART + blockIdx.x] = S1;
    part[2 * NB_PART + blockIdx.x] = Q0;
    part[3 * NB_PART + blockIdx.x] = Q1;
  }
}

__global__ __launch_bounds__(256) void finalize_stats_kernel(const float* __restrict__ sums,
                                                             const float* __restrict__ sqs,
                                                             int nb, float count,
                                                             float* __restrict__ out2) {
  __shared__ float red[256];
  float s = (threadIdx.x < nb) ? sums[threadIdx.x] : 0.f;
  float q = (threadIdx.x < nb) ? sqs[threadIdx.x] : 0.f;
  float S = block_reduce_sum(s, red);
  float Q = block_reduce_sum(q, red);
  if (threadIdx.x == 0) {
    float mu = S / count;
    float var = Q / count - mu * mu;    // biased batch variance
    out2[0] = mu;
    out2[1] = rsqrtf(var + BN_EPS);
  }
}

__global__ __launch_bounds__(256) void xg1_kernel(const float* __restrict__ g,
                                                  const float* __restrict__ mirna,
                                                  const float* __restrict__ gw1,
                                                  const float* __restrict__ gb1,
                                                  float* __restrict__ xg1) {
  __shared__ float red[256];
  const int m = blockIdx.x;
  const int dv = dv_row(m);
  float s0 = 0.f, s1 = 0.f;
  for (int w = threadIdx.x; w < F_OUT; w += 256) {
    s0 += g[(long long)dv * F_OUT + w];
    s1 += mirna[(long long)m * F_OUT + w];
  }
  float S0 = block_reduce_sum(s0, red);
  float S1 = block_reduce_sum(s1, red);
  if (threadIdx.x == 0) {
    float gp0 = S0 / (float)F_OUT, gp1 = S1 / (float)F_OUT;
    xg1[m] = gw1[0] * gp0 + gw1[1] * gp1 + gb1[0];
  }
}

__global__ __launch_bounds__(256) void xg_stats_kernel(const float* __restrict__ xg1,
                                                       const float* __restrict__ gw2,
                                                       const float* __restrict__ gb2,
                                                       float* __restrict__ stg) {
  __shared__ float red[256];
  __shared__ float mu1s, rs1s;
  float s = 0.f, q = 0.f;
  for (int m = threadIdx.x; m < M_ROWS; m += 256) {
    float v = xg1[m];
    s += v; q += v * v;
  }
  float S = block_reduce_sum(s, red);
  float Q = block_reduce_sum(q, red);
  if (threadIdx.x == 0) {
    float mu = S / (float)M_ROWS;
    float var = Q / (float)M_ROWS - mu * mu;
    mu1s = mu; rs1s = rsqrtf(var + BN_EPS);
  }
  __syncthreads();
  const float mu1 = mu1s, rs1 = rs1s;
  for (int c = 0; c < 2; ++c) {
    s = 0.f; q = 0.f;
    for (int m = threadIdx.x; m < M_ROWS; m += 256) {
      float r = fmaxf((xg1[m] - mu1) * rs1, 0.f);
      float v = gw2[c] * r + gb2[c];
      s += v; q += v * v;
    }
    float Sc = block_reduce_sum(s, red);
    float Qc = block_reduce_sum(q, red);
    if (threadIdx.x == 0) {
      float mu = Sc / (float)M_ROWS;
      float var = Qc / (float)M_ROWS - mu * mu;
      stg[2 + 2 * c] = mu;
      stg[3 + 2 * c] = rsqrtf(var + BN_EPS);
    }
    __syncthreads();
  }
  if (threadIdx.x == 0) { stg[0] = mu1; stg[1] = rs1; }
}

__global__ void fuse_kernel(const float* __restrict__ g, const float* __restrict__ mirna,
                            const float* __restrict__ t0, const float* __restrict__ lw2,
                            const float* __restrict__ lb2, const float* __restrict__ xg1,
                            const float* __restrict__ gw2, const float* __restrict__ gb2,
                            const float* __restrict__ st0, const float* __restrict__ st2,
                            const float* __restrict__ stg, half_t* __restrict__ fused_h) {
  const long long total = (long long)M_ROWS * KPAD;
  for (long long t = blockIdx.x * (long long)blockDim.x + threadIdx.x; t < total;
       t += (long long)gridDim.x * blockDim.x) {
    int m = (int)(t / KPAD), w = (int)(t % KPAD);
    if (w >= F_OUT) { fused_h[t] = (half_t)0.f; continue; }
    float x0 = g[(long long)dv_row(m) * F_OUT + w];
    float x1 = mirna[(long long)m * F_OUT + w];
    float r = fmaxf((t0[(long long)m * F_OUT + w] - st0[0]) * st0[1], 0.f);
    float xl0 = ((lw2[0] * r + lb2[0]) - st2[0]) * st2[1];
    float xl1 = ((lw2[1] * r + lb2[1]) - st2[2]) * st2[3];
    float rg = fmaxf((xg1[m] - stg[0]) * stg[1], 0.f);
    float xgc0 = ((gw2[0] * rg + gb2[0]) - stg[2]) * stg[3];
    float xgc1 = ((gw2[1] * rg + gb2[1]) - stg[4]) * stg[5];
    float w0 = 1.f / (1.f + __expf(-(xl0 + xgc0)));
    float w1 = 1.f / (1.f + __expf(-(xl1 + xgc1)));
    fused_h[t] = (half_t)(0.5f * (x0 * w0 + x1 * w1));
  }
}

// ---------------- fused 4-layer MLP scorer (WMMA, LDS-staged) ----------------
__global__ __launch_bounds__(256) void mlp_kernel(const half_t* __restrict__ fused_h,
                                                  const int* __restrict__ train,
                                                  const int* __restrict__ test,
                                                  const half_t* __restrict__ W1p,
                                                  const float* __restrict__ b1,
                                                  const half_t* __restrict__ W2p,
                                                  const float* __restrict__ b2,
                                                  const half_t* __restrict__ W3p,
                                                  const float* __restrict__ b3,
                                                  const float* __restrict__ W4,
                                                  const float* __restrict__ b4,
                                                  float* __restrict__ out) {
  __shared__ half_t h1[16][N1];
  __shared__ half_t h2[16][N2];
  __shared__ float  h3[16][N3];
  __shared__ int p0s[16], p1s[16];

  const int tile = blockIdx.x;
  if (threadIdx.x < 16) {
    int row = tile * 16 + threadIdx.x;
    int p0, p1;
    if (row < NTRAIN) { p0 = train[2 * row]; p1 = train[2 * row + 1]; }
    else { int r2 = row - NTRAIN; p0 = test[2 * r2]; p1 = test[2 * r2 + 1]; }
    p0s[threadIdx.x] = p0;
    p1s[threadIdx.x] = p1;
  }
  __syncthreads();

  const int wv = threadIdx.x >> 5, lane = threadIdx.x & 31;
  const int l16 = lane & 15, lh = lane >> 4;
  const long long base0 = (long long)p0s[l16] * KPAD;
  const long long base1 = (long long)p1s[l16] * KPAD;

  // ---- layer 1: [16 x 1856] x [1856 x 1024], 58 K-steps, 8 N-tiles / wave ----
  v8f acc[8];
#pragma unroll
  for (int i = 0; i < 8; ++i) acc[i] = zero8();
  {
    const half_t* bp0 = W1p + (((long long)(wv * 8) * L1_KS) * 32 + lane) * 16;
    for (int ks = 0; ks < L1_KS; ++ks) {
      const int off0 = ks * 32 + lh * 8;
      const half_t* src = (off0 < KPAD) ? (fused_h + base0 + off0)
                                        : (fused_h + base1 + (off0 - KPAD));
      v8h lo = *(const v8h*)src;
      v8h hi = *(const v8h*)(src + 16);
      v16h a = join16(lo, hi);
#pragma unroll
      for (int i = 0; i < 8; ++i) {
        const half_t* bp = bp0 + ((long long)i * L1_KS + ks) * 512;
        v16h bfr = *(const v16h*)bp;
        __builtin_prefetch(bp + 512, 0, 1);
        acc[i] = wmma16(a, bfr, acc[i]);
      }
    }
  }
#pragma unroll
  for (int i = 0; i < 8; ++i) {
    const int n = (wv * 8 + i) * 16 + l16;
#pragma unroll
    for (int r = 0; r < 8; ++r) h1[r + lh * 8][n] = (half_t)(acc[i][r] + b1[n]);
  }
  __syncthreads();

  // ---- layer 2: [16 x 1024] x [1024 x 512], 32 K-steps, 4 N-tiles / wave ----
  v8f acc2[4];
#pragma unroll
  for (int i = 0; i < 4; ++i) acc2[i] = zero8();
  {
    const half_t* bp0 = W2p + (((long long)(wv * 4) * L2_KS) * 32 + lane) * 16;
    for (int ks = 0; ks < L2_KS; ++ks) {
      const int off0 = ks * 32 + lh * 8;
      v8h lo = *(const v8h*)&h1[l16][off0];
      v8h hi = *(const v8h*)&h1[l16][off0 + 16];
      v16h a = join16(lo, hi);
#pragma unroll
      for (int i = 0; i < 4; ++i) {
        const half_t* bp = bp0 + ((long long)i * L2_KS + ks) * 512;
        v16h bfr = *(const v16h*)bp;
        acc2[i] = wmma16(a, bfr, acc2[i]);
      }
    }
  }
#pragma unroll
  for (int i = 0; i < 4; ++i) {
    const int n = (wv * 4 + i) * 16 + l16;
#pragma unroll
    for (int r = 0; r < 8; ++r) h2[r + lh * 8][n] = (half_t)(acc2[i][r] + b2[n]);
  }
  __syncthreads();

  // ---- layer 3: [16 x 512] x [512 x 64], 16 K-steps, waves 0..3 ----
  if (wv < 4) {
    v8f acc3 = zero8();
    const int n = wv * 16 + l16;
    const half_t* bp = W3p + (((long long)wv * L3_KS) * 32 + lane) * 16;
    for (int ks = 0; ks < L3_KS; ++ks) {
      const int off0 = ks * 32 + lh * 8;
      v8h lo = *(const v8h*)&h2[l16][off0];
      v8h hi = *(const v8h*)&h2[l16][off0 + 16];
      v16h a = join16(lo, hi);
      v16h bfr = *(const v16h*)bp;
      bp += 512;
      acc3 = wmma16(a, bfr, acc3);
    }
#pragma unroll
    for (int r = 0; r < 8; ++r) h3[r + lh * 8][n] = acc3[r] + b3[n];
  }
  __syncthreads();

  // ---- layer 4 + sigmoid: 16 rows, 64-wide dot ----
  if (threadIdx.x < 16) {
    float s = b4[0];
#pragma unroll
    for (int j = 0; j < N3; ++j) s += h3[threadIdx.x][j] * W4[j];
    int row = tile * 16 + threadIdx.x;
    if (row < BTOT) out[row] = 1.f / (1.f + __expf(-s));
  }
}

// ---------------- host orchestration ----------------
extern "C" void kernel_launch(void* const* d_in, const int* in_sizes, int n_in,
                              void* d_out, int out_size, void* d_ws, size_t ws_size,
                              hipStream_t stream) {
  (void)in_sizes; (void)n_in; (void)out_size; (void)ws_size;
  const float* x_feat  = (const float*)d_in[0];
  const float* mirna   = (const float*)d_in[1];
  const float* Wg      = (const float*)d_in[2];
  const float* att_src = (const float*)d_in[3];
  const float* att_dst = (const float*)d_in[4];
  const float* gat_b   = (const float*)d_in[5];
  const float* lw1     = (const float*)d_in[6];
  const float* lb1     = (const float*)d_in[7];
  const float* lw2     = (const float*)d_in[8];
  const float* lb2     = (const float*)d_in[9];
  const float* gw1     = (const float*)d_in[10];
  const float* gb1     = (const float*)d_in[11];
  const float* gw2     = (const float*)d_in[12];
  const float* gb2     = (const float*)d_in[13];
  const float* W1      = (const float*)d_in[14];
  const float* b1      = (const float*)d_in[15];
  const float* W2      = (const float*)d_in[16];
  const float* b2      = (const float*)d_in[17];
  const float* W3      = (const float*)d_in[18];
  const float* b3      = (const float*)d_in[19];
  const float* W4      = (const float*)d_in[20];
  const float* b4      = (const float*)d_in[21];
  const int* edge_index = (const int*)d_in[22];
  const int* train      = (const int*)d_in[23];
  const int* test       = (const int*)d_in[24];
  float* out = (float*)d_out;

  // workspace carve-out (~40 MB total)
  char* ws = (char*)d_ws;
  size_t off = 0;
  auto take = [&](size_t bytes) -> char* {
    char* p = ws + off;
    off = (off + bytes + 255) & ~(size_t)255;
    return p;
  };
  float*    H     = (float*)take(sizeof(float) * (size_t)N_NODES * F_OUT);
  float*    G     = (float*)take(sizeof(float) * (size_t)N_NODES * F_OUT);
  float*    a_s   = (float*)take(sizeof(float) * N_NODES);
  float*    a_d   = (float*)take(sizeof(float) * N_NODES);
  unsigned* emax  = (unsigned*)take(sizeof(unsigned) * N_NODES);
  float*    denom = (float*)take(sizeof(float) * N_NODES);
  float*    elog  = (float*)take(sizeof(float) * TOT_E);
  float*    eex   = (float*)take(sizeof(float) * TOT_E);
  float*    t0    = (float*)take(sizeof(float) * (size_t)M_ROWS * F_OUT);
  float*    xg1   = (float*)take(sizeof(float) * M_ROWS);
  float*    part0 = (float*)take(sizeof(float) * 2 * NB_PART);
  float*    part2 = (float*)take(sizeof(float) * 4 * NB_PART);
  float*    st0   = (float*)take(sizeof(float) * 2);
  float*    st2   = (float*)take(sizeof(float) * 4);
  float*    stg   = (float*)take(sizeof(float) * 6);
  half_t*   fusedh= (half_t*)take(sizeof(half_t) * (size_t)M_ROWS * KPAD);
  half_t*   Xh    = (half_t*)take(sizeof(half_t) * (size_t)G_MP * G_KP);
  half_t*   Wgp   = (half_t*)take(sizeof(half_t) * (size_t)G_NT * G_KS * 512);
  half_t*   W1p   = (half_t*)take(sizeof(half_t) * (size_t)L1_NT * L1_KS * 512);
  half_t*   W2p   = (half_t*)take(sizeof(half_t) * (size_t)L2_NT * L2_KS * 512);
  half_t*   W3p   = (half_t*)take(sizeof(half_t) * (size_t)L3_NT * L3_KS * 512);

  // prep: pad/convert A, fragment-swizzle all B operands
  pack_xh_kernel<<<2048, 256, 0, stream>>>(x_feat, Xh);
  pack_b_kernel<<<1024, 256, 0, stream>>>(Wg, Wgp, N_NODES, F_OUT, G_KS, G_NT, 0);
  pack_b_kernel<<<1024, 256, 0, stream>>>(W1, W1p, 1802, N1, L1_KS, L1_NT, 1);
  pack_b_kernel<<<512, 256, 0, stream>>>(W2, W2p, N1, N2, L2_KS, L2_NT, 0);
  pack_b_kernel<<<64, 256, 0, stream>>>(W3, W3p, N2, N3, L3_KS, L3_NT, 0);

  // GAT
  gat_gemm_kernel<<<G_MP / 16, 256, 0, stream>>>(Xh, Wgp, H);
  attn_scores_kernel<<<N_NODES, 256, 0, stream>>>(H, att_src, att_dst, a_s, a_d);
  gat_init_kernel<<<2048, 256, 0, stream>>>(emax, denom, G);
  edge_logit_kernel<<<512, 256, 0, stream>>>(edge_index, a_s, a_d, elog, emax);
  edge_exp_kernel<<<512, 256, 0, stream>>>(edge_index, elog, emax, eex, denom);
  edge_alpha_kernel<<<512, 256, 0, stream>>>(edge_index, denom, eex);
  edge_scatter_kernel<<<4096, 256, 0, stream>>>(edge_index, eex, H, G);
  gat_bias_relu_kernel<<<2048, 256, 0, stream>>>(G, gat_b);

  // MS_CAM batch-norm statistics + fusion
  t0_stats_kernel<<<NB_PART, 256, 0, stream>>>(G, mirna, lw1, lb1, t0, part0);
  finalize_stats_kernel<<<1, 256, 0, stream>>>(part0, part0 + NB_PART, NB_PART,
                                               (float)((long long)M_ROWS * F_OUT), st0);
  t2_stats_kernel<<<NB_PART, 256, 0, stream>>>(t0, lw2, lb2, st0, part2);
  finalize_stats_kernel<<<1, 256, 0, stream>>>(part2, part2 + 2 * NB_PART, NB_PART,
                                               (float)((long long)M_ROWS * F_OUT), st2);
  finalize_stats_kernel<<<1, 256, 0, stream>>>(part2 + NB_PART, part2 + 3 * NB_PART, NB_PART,
                                               (float)((long long)M_ROWS * F_OUT), st2 + 2);
  xg1_kernel<<<M_ROWS, 256, 0, stream>>>(G, mirna, gw1, gb1, xg1);
  xg_stats_kernel<<<1, 256, 0, stream>>>(xg1, gw2, gb2, stg);
  fuse_kernel<<<1024, 256, 0, stream>>>(G, mirna, t0, lw2, lb2, xg1, gw2, gb2,
                                        st0, st2, stg, fusedh);

  // fused MLP scorer: 130000 rows / 16 per workgroup = 8125 blocks
  mlp_kernel<<<BTOT / 16, 256, 0, stream>>>(fusedh, train, test, W1p, b1, W2p, b2,
                                            W3p, b3, W4, b4, out);
}
